// Correction_62534723829962
// MI455X (gfx1250) — matrix operand order repaired and checked by reading, not compiled, not run on hardware
//
#include <hip/hip_runtime.h>
#include <hip/hip_bf16.h>

#define L_  4
#define C_  256
#define H_  128
#define W_  256
#define CD_ 128
#define HW_ (H_*W_)
#define KP_ 256

typedef __attribute__((ext_vector_type(16))) __bf16 v16bf;
typedef __attribute__((ext_vector_type(8)))  float  v8f;

union Frag { uint4 q[2]; v16bf v; };

// ---------------- one-time repack: weights -> bf16 [d][K] layouts, zero page ----------------
__global__ void k_repack(const float* __restrict__ wD, const float* __restrict__ wP,
                         const float* __restrict__ wS,
                         __bf16* __restrict__ wbfD, __bf16* __restrict__ wbfP,
                         __bf16* __restrict__ wbfS, float* __restrict__ zeros) {
  int i = blockIdx.x * 256 + threadIdx.x;
  if (i < 294912) {                       // convD: K = tap*256 + c
    int d = i / 2304, r = i % 2304;
    int tap = r >> 8, c = r & 255;
    wbfD[i] = (__bf16)wD[(size_t)(d * 256 + c) * 9 + tap];
  }
  if (i < 49152) {                        // proj: K = tap*128 + c
    int d = i / 384, r = i % 384;
    int tap = r >> 7, c = r & 127;
    wbfP[i] = (__bf16)wP[(size_t)(d * 128 + c) * 3 + tap];
  }
  if (i < 16384) wbfS[i] = (__bf16)wS[i]; // projS already [d][c]
  if (i < 64) zeros[i] = 0.f;
}

// ---------------- scores conv (1 out ch, 3x3, sigmoid) ----------------
__global__ void k_scores_conv(const float* __restrict__ feats,
                              const float* __restrict__ w,
                              const float* __restrict__ b,
                              float* __restrict__ S) {
  __shared__ float sw[2304];
  __shared__ float srow[3][W_];
  int t = threadIdx.x;
  int ly = blockIdx.x;
  int l = ly / H_, y = ly % H_;
  for (int i = t; i < 2304; i += W_) sw[i] = w[i];
  float acc = b[0];
  const float* fl = feats + (size_t)l * C_ * HW_;
  for (int c = 0; c < C_; ++c) {
    __syncthreads();
#pragma unroll
    for (int r = 0; r < 3; ++r) {
      int yy = y + r - 1;
      srow[r][t] = (yy >= 0 && yy < H_) ? fl[(size_t)c * HW_ + yy * W_ + t] : 0.f;
    }
    if (c + 1 < C_) __builtin_prefetch(fl + (size_t)(c + 1) * HW_ + y * W_ + t, 0, 1);
    __syncthreads();
#pragma unroll
    for (int r = 0; r < 3; ++r)
#pragma unroll
      for (int dx = 0; dx < 3; ++dx) {
        int xx = t + dx - 1;
        float v = (xx >= 0 && xx < W_) ? srow[r][xx] : 0.f;
        acc = fmaf(v, sw[c * 9 + r * 3 + dx], acc);
      }
  }
  S[(size_t)ly * W_ + t] = 1.f / (1.f + __expf(-acc));
}

// ---------------- separable 9-tap max pools ----------------
__global__ void k_pool_x(const float* __restrict__ in, float* __restrict__ out) {
  int t = threadIdx.x; int ly = blockIdx.x;
  const float* row = in + (size_t)ly * W_;
  float m = -__builtin_inff();
#pragma unroll
  for (int d = -4; d <= 4; ++d) { int x = t + d; if (x >= 0 && x < W_) m = fmaxf(m, row[x]); }
  out[(size_t)ly * W_ + t] = m;
}
__global__ void k_pool_y(const float* __restrict__ in, float* __restrict__ out) {
  int t = threadIdx.x; int ly = blockIdx.x; int l = ly / H_, y = ly % H_;
  float m = -__builtin_inff();
#pragma unroll
  for (int d = -4; d <= 4; ++d) {
    int yy = y + d;
    if (yy >= 0 && yy < H_) m = fmaxf(m, in[((size_t)l * H_ + yy) * W_ + t]);
  }
  out[(size_t)ly * W_ + t] = m;
}

// ---------------- NMS elementwise steps ----------------
__global__ void k_eq_mask(const float* S, const float* P, float* M) {
  size_t i = (size_t)blockIdx.x * blockDim.x + threadIdx.x;
  M[i] = (S[i] == P[i]) ? 1.f : 0.f;
}
__global__ void k_supp_scores(const float* P, const float* S, float* SS) {
  size_t i = (size_t)blockIdx.x * blockDim.x + threadIdx.x;
  SS[i] = (P[i] > 0.f) ? 0.f : S[i];
}
__global__ void k_mask_update(float* M, const float* SS, const float* P2, const float* P) {
  size_t i = (size_t)blockIdx.x * blockDim.x + threadIdx.x;
  bool nm = (SS[i] == P2[i]) && !(P[i] > 0.f);
  M[i] = ((M[i] != 0.f) || nm) ? 1.f : 0.f;
}
__global__ void k_apply_mask(const float* M, const float* S, float* NM) {
  size_t i = (size_t)blockIdx.x * blockDim.x + threadIdx.x;
  NM[i] = (M[i] != 0.f) ? S[i] : 0.f;
}

// ---------------- exact iterative top-256 (block per layer, 1024 thr) ----------------
__global__ void k_topk(const float* __restrict__ NM, int* __restrict__ topk) {
  int l = blockIdx.x;
  int t = threadIdx.x;
  float v[32];
  int base = t * 32;
  const float* src = NM + (size_t)l * HW_;
#pragma unroll
  for (int i = 0; i < 32; ++i) v[i] = src[base + i];
  __shared__ float skey[1024];
  __shared__ int   sidx[1024];
  for (int r = 0; r < KP_; ++r) {
    float bv = v[0]; int bi = 0;
#pragma unroll
    for (int i = 1; i < 32; ++i) if (v[i] > bv) { bv = v[i]; bi = i; }
    skey[t] = bv; sidx[t] = base + bi;
    __syncthreads();
    for (int s = 512; s > 0; s >>= 1) {
      if (t < s) {
        if (skey[t + s] > skey[t] ||
            (skey[t + s] == skey[t] && sidx[t + s] < sidx[t])) {
          skey[t] = skey[t + s]; sidx[t] = sidx[t + s];
        }
      }
      __syncthreads();
    }
    int win = sidx[0];
    if (t == 0) topk[l * KP_ + r] = win;
    if (win >= base && win < base + 32) v[win - base] = -__builtin_inff();
    __syncthreads();
  }
}

// async stage one 3x3 tap (16 kp x 256 ch fp32) into an LDS buffer
__device__ __forceinline__ void stage_tap(const float* __restrict__ fl,
                                          const float* __restrict__ zeros,
                                          float* AfBuf, int skp, int cg,
                                          int ky0, int kx0, int tap) {
  int dy = tap / 3 - 1, dx = tap % 3 - 1;
  int yy = ky0 + dy, xx = kx0 + dx;
  bool valid = (yy >= 0 && yy < H_ && xx >= 0 && xx < W_);
  const float* base = fl + (size_t)(yy * W_ + xx);
#pragma unroll
  for (int i = 0; i < 16; ++i) {
    int c = cg + 16 * i;
    const float* g = valid ? (base + (size_t)c * HW_) : zeros;
    uint32_t loff = (uint32_t)(uintptr_t)(&AfBuf[skp * 256 + c]); // LDS off = addr[31:0]
    asm volatile("global_load_async_to_lds_b32 %0, %1, off"
                 :: "v"(loff), "v"(g) : "memory");
  }
}

// ---- desc conv at keypoints: double-buffered async-to-LDS A, bf16 WMMA, K=2304 ----
__global__ void k_desc_conv(const float* __restrict__ feats,
                            const __bf16* __restrict__ wbfD, // [d][tap*256+c]
                            const float* __restrict__ bD,
                            const int* __restrict__ topk,
                            const float* __restrict__ zeros,
                            float* __restrict__ descbuf) {
  __shared__ __align__(16) float Af[2][16 * 256];   // 2 x 16KB double buffer
  int blk = blockIdx.x;
  int l = blk >> 4, mtile = blk & 15;
  int t = threadIdx.x;
  int wave = t >> 5, lane = t & 31;
  int half = lane >> 4, lm = lane & 15;
  int dch = wave * 16 + lm;

  int skp = t & 15;      // staged keypoint
  int cg  = t >> 4;      // staged channel group
  int kidx = topk[l * KP_ + mtile * 16 + skp];
  int ky0 = kidx >> 8, kx0 = kidx & 255;
  const float* fl = feats + (size_t)l * C_ * HW_;
  const __bf16* wrow = wbfD + (size_t)dch * 2304;

  v8f acc = {0.f,0.f,0.f,0.f,0.f,0.f,0.f,0.f};

  stage_tap(fl, zeros, &Af[0][0], skp, cg, ky0, kx0, 0);
  asm volatile("s_wait_asynccnt 0" ::: "memory");
  __syncthreads();

  for (int tap = 0; tap < 9; ++tap) {
    // overlap: DMA next tap into the other buffer while computing this one
    if (tap + 1 < 9)
      stage_tap(fl, zeros, &Af[(tap + 1) & 1][0], skp, cg, ky0, kx0, tap + 1);

    const float* cur = &Af[tap & 1][0];
    const __bf16* wtap = wrow + (size_t)tap * 256;
    Frag Bcur;
    {
      const uint4* bp = (const uint4*)(wtap + 8 * half);
      Bcur.q[0] = bp[0]; Bcur.q[1] = bp[2];
    }
#pragma unroll
    for (int ks = 0; ks < 8; ++ks) {
      Frag Bnext;
      if (ks < 7) {  // pipeline B one step ahead
        const uint4* bp = (const uint4*)(wtap + (ks + 1) * 32 + 8 * half);
        Bnext.q[0] = bp[0]; Bnext.q[1] = bp[2];
      }
      int kb = ks * 32;
      const float4* r0 = (const float4*)&cur[lm * 256 + kb + 8 * half];
      const float4* r1 = (const float4*)&cur[lm * 256 + kb + 16 + 8 * half];
      float f0[8], f1[8];
      *(float4*)&f0[0] = r0[0]; *(float4*)&f0[4] = r0[1];
      *(float4*)&f1[0] = r1[0]; *(float4*)&f1[4] = r1[1];
      v16bf a;
#pragma unroll
      for (int e = 0; e < 8; ++e) { a[e] = (__bf16)f0[e]; a[e + 8] = (__bf16)f1[e]; }
      acc = __builtin_amdgcn_wmma_f32_16x16x32_bf16(false, a, false, Bcur.v,
                                                    (short)0, acc, false, false);
      if (ks < 7) Bcur = Bnext;
    }
    if (tap + 1 < 9) {
      asm volatile("s_wait_asynccnt 0" ::: "memory");
      __syncthreads();
    }
  }
  float bias = bD[dch];
#pragma unroll
  for (int r = 0; r < 8; ++r) {
    int kp = mtile * 16 + r + 8 * half;
    descbuf[((size_t)l * CD_ + dch) * KP_ + kp] = fmaxf(acc[r] + bias, 0.f);
  }
}

// ---------------- L2 normalize over channels (block per (l,k)) ----------------
__global__ void k_normalize(float* __restrict__ descbuf) {
  int lk = blockIdx.x;
  int l = lk >> 8, k = lk & 255;
  int d = threadIdx.x;
  float* p = descbuf + (size_t)l * CD_ * KP_ + k;
  float v = p[(size_t)d * KP_];
  __shared__ float red[CD_];
  red[d] = v * v;
  __syncthreads();
  for (int s = 64; s > 0; s >>= 1) { if (d < s) red[d] += red[d + s]; __syncthreads(); }
  float norm = fmaxf(sqrtf(red[0]), 1e-12f);
  p[(size_t)d * KP_] = v / norm;
}

// ---- per-keypoint 4x4 attention; emits bf16 transposed [l*KP+k][c] for proj WMMA ----
__global__ void k_attention(const float* __restrict__ descbuf, __bf16* __restrict__ d1bf) {
  int k = blockIdx.x;
  int c = threadIdx.x;
  __shared__ float q[L_][CD_];
  __shared__ float attn[L_][L_];
  __shared__ float prob[L_][L_];
#pragma unroll
  for (int l = 0; l < L_; ++l) q[l][c] = descbuf[((size_t)l * CD_ + c) * KP_ + k];
  __syncthreads();
  if (c < 16) {
    int n = c >> 2, m = c & 3;
    float s = 0.f;
    for (int cc = 0; cc < CD_; ++cc) s += q[n][cc] * q[m][cc];
    attn[n][m] = s * 0.08838834764831845f; // 1/sqrt(128)
  }
  __syncthreads();
  if (c < 4) {
    int n = c;
    float mx = attn[n][0];
    for (int m = 1; m < 4; ++m) mx = fmaxf(mx, attn[n][m]);
    float e0[4]; float den = 0.f;
    for (int m = 0; m < 4; ++m) { e0[m] = __expf(attn[n][m] - mx); den += e0[m]; }
    for (int m = 0; m < 4; ++m) prob[n][m] = e0[m] / den;
  }
  __syncthreads();
#pragma unroll
  for (int l = 0; l < L_; ++l) {
    float msg = 0.f;
#pragma unroll
    for (int m = 0; m < L_; ++m) msg += prob[l][m] * q[m][c];
    d1bf[((size_t)l * KP_ + k) * CD_ + c] = (__bf16)(2.f * q[l][c] + msg);
  }
}

// ---- proj conv1d k=3 pad1 relu: WMMA M=256 N=128 K=384, contiguous bf16 operands ----
__global__ void k_proj_conv(const __bf16* __restrict__ d1bf, // [l*KP+k][CD]
                            const __bf16* __restrict__ wbfP, // [d][tap*128+c]
                            const float* __restrict__ b,
                            float* __restrict__ out,         // [l][d][k]
                            __bf16* __restrict__ pkbf) {     // [l*KP+k][d]
  int blk = blockIdx.x; int l = blk >> 4, mtile = blk & 15;
  int wave = threadIdx.x >> 5, lane = threadIdx.x & 31;
  int half = lane >> 4, lm = lane & 15;
  int kpos = mtile * 16 + lm;
  int dch = wave * 16 + lm;
  const __bf16* wrow = wbfP + (size_t)dch * 384;
  v8f acc = {0.f,0.f,0.f,0.f,0.f,0.f,0.f,0.f};
  uint4 z = {0u,0u,0u,0u};
#pragma unroll
  for (int tap = 0; tap < 3; ++tap) {
    int pos = kpos + tap - 1;
    bool valid = (pos >= 0 && pos < KP_);
    const __bf16* arow = d1bf + ((size_t)l * KP_ + (valid ? pos : 0)) * CD_;
#pragma unroll
    for (int ks = 0; ks < 4; ++ks) {
      int kb = ks * 32;
      Frag A, B;
      A.q[0] = valid ? *(const uint4*)(arow + kb + 8 * half)      : z;
      A.q[1] = valid ? *(const uint4*)(arow + kb + 16 + 8 * half) : z;
      const uint4* bp = (const uint4*)(wrow + tap * 128 + kb + 8 * half);
      B.q[0] = bp[0]; B.q[1] = bp[2];
      acc = __builtin_amdgcn_wmma_f32_16x16x32_bf16(false, A.v, false, B.v,
                                                    (short)0, acc, false, false);
    }
  }
  float bias = b[dch];
#pragma unroll
  for (int r = 0; r < 8; ++r) {
    int kp = mtile * 16 + r + 8 * half;
    float v = fmaxf(acc[r] + bias, 0.f);
    out[((size_t)l * CD_ + dch) * KP_ + kp] = v;
    pkbf[((size_t)l * KP_ + kp) * CD_ + dch] = (__bf16)v;
  }
}

// ---- projS conv1d k=1: WMMA M=256 N=128 K=128 ----
__global__ void k_projS_conv(const __bf16* __restrict__ pkbf, // [l*KP+k][d]
                             const __bf16* __restrict__ wbfS, // [d][c]
                             const float* __restrict__ b,
                             float* __restrict__ sout) {
  int blk = blockIdx.x; int l = blk >> 4, mtile = blk & 15;
  int wave = threadIdx.x >> 5, lane = threadIdx.x & 31;
  int half = lane >> 4, lm = lane & 15;
  int kpos = mtile * 16 + lm;
  int dch = wave * 16 + lm;
  const __bf16* arow = pkbf + ((size_t)l * KP_ + kpos) * CD_;
  const __bf16* wrow = wbfS + (size_t)dch * CD_;
  v8f acc = {0.f,0.f,0.f,0.f,0.f,0.f,0.f,0.f};
#pragma unroll
  for (int ks = 0; ks < 4; ++ks) {
    int kb = ks * 32;
    Frag A, B;
    A.q[0] = *(const uint4*)(arow + kb + 8 * half);
    A.q[1] = *(const uint4*)(arow + kb + 16 + 8 * half);
    const uint4* bp = (const uint4*)(wrow + kb + 8 * half);
    B.q[0] = bp[0]; B.q[1] = bp[2];
    acc = __builtin_amdgcn_wmma_f32_16x16x32_bf16(false, A.v, false, B.v,
                                                  (short)0, acc, false, false);
  }
  float bias = b[dch];
#pragma unroll
  for (int r = 0; r < 8; ++r) {
    int kp = mtile * 16 + r + 8 * half;
    sout[((size_t)l * CD_ + dch) * KP_ + kp] = acc[r] + bias;
  }
}

// ---------------- per-(l,d) bitonic sort descending by s ----------------
__global__ void k_sort(const float* __restrict__ sout, const float* __restrict__ pin,
                       float* __restrict__ sorted) {
  int ld = blockIdx.x;
  int t = threadIdx.x;
  __shared__ float key[KP_];
  __shared__ float val[KP_];
  key[t] = sout[(size_t)ld * KP_ + t];
  val[t] = pin[(size_t)ld * KP_ + t];
  __syncthreads();
  for (int size = 2; size <= KP_; size <<= 1) {
    for (int stride = size >> 1; stride > 0; stride >>= 1) {
      int partner = t ^ stride;
      if (partner > t) {
        bool up = ((t & size) == 0);
        float k0 = key[t], k1 = key[partner];
        bool swap = up ? (k0 < k1) : (k0 > k1);
        if (swap) {
          float v0 = val[t], v1 = val[partner];
          key[t] = k1; key[partner] = k0;
          val[t] = v1; val[partner] = v0;
        }
      }
      __syncthreads();
    }
  }
  sorted[(size_t)ld * KP_ + t] = val[t];
}

// ---------------- final 3x128 k=1 conv ----------------
__global__ void k_final(const float* __restrict__ sorted,
                        const float* __restrict__ w, const float* __restrict__ b,
                        float* __restrict__ out3) {
  int l = blockIdx.x; int k = threadIdx.x;
#pragma unroll
  for (int j = 0; j < 3; ++j) {
    float acc = b[j];
    for (int c = 0; c < CD_; ++c)
      acc = fmaf(w[j * CD_ + c], sorted[((size_t)l * CD_ + c) * KP_ + k], acc);
    out3[((size_t)l * 3 + j) * KP_ + k] = acc;
  }
}

// ---------------- min |out3 - out3[0]| over k ----------------
__global__ void k_mindesc(const float* __restrict__ out3, float* __restrict__ md) {
  int lj = blockIdx.x;
  int l = lj / 3, j = lj % 3;
  int t = threadIdx.x;
  float diff = out3[((size_t)l * 3 + j) * KP_ + t] - out3[(size_t)j * KP_ + t];
  __shared__ float av[KP_];
  __shared__ int   ai[KP_];
  av[t] = fabsf(diff); ai[t] = t;
  __syncthreads();
  for (int s = 128; s > 0; s >>= 1) {
    if (t < s) {
      if (av[t + s] < av[t] || (av[t + s] == av[t] && ai[t + s] < ai[t])) {
        av[t] = av[t + s]; ai[t] = ai[t + s];
      }
    }
    __syncthreads();
  }
  if (t == 0) md[lj] = av[0];
}

// ---------------- build affine matrices ----------------
__global__ void k_theta(const float* __restrict__ md, float* __restrict__ theta) {
  int l = threadIdx.x;
  if (l < L_) {
    float th = md[l * 3 + 2], tx = md[l * 3 + 0], ty = md[l * 3 + 1];
    float ct = cosf(th), st = sinf(th);
    theta[l * 6 + 0] = ct;  theta[l * 6 + 1] = -st; theta[l * 6 + 2] = tx;
    theta[l * 6 + 3] = st;  theta[l * 6 + 4] = ct;  theta[l * 6 + 5] = ty;
  }
}

// ---------------- bilinear grid sample ----------------
__global__ void k_grid_sample(const float* __restrict__ feats,
                              const float* __restrict__ theta,
                              float* __restrict__ out) {
  int ly = blockIdx.x; int l = ly / H_, y = ly % H_;
  int x = threadIdx.x;
  float X = (x + 0.5f) * (2.f / W_) - 1.f;
  float Y = (y + 0.5f) * (2.f / H_) - 1.f;
  const float* th = theta + l * 6;
  float gx = th[0] * X + th[1] * Y + th[2];
  float gy = th[3] * X + th[4] * Y + th[5];
  float ix = ((gx + 1.f) * W_ - 1.f) * 0.5f;
  float iy = ((gy + 1.f) * H_ - 1.f) * 0.5f;
  float x0f = floorf(ix), y0f = floorf(iy);
  int x0 = (int)x0f, y0 = (int)y0f;
  int x1 = x0 + 1, y1 = y0 + 1;
  float wa = (x1 - ix) * (y1 - iy);
  float wb = (ix - x0f) * (y1 - iy);
  float wc = (x1 - ix) * (iy - y0f);
  float wd = (ix - x0f) * (iy - y0f);
  bool va = (x0 >= 0 && x0 < W_ && y0 >= 0 && y0 < H_);
  bool vb = (x1 >= 0 && x1 < W_ && y0 >= 0 && y0 < H_);
  bool vc = (x0 >= 0 && x0 < W_ && y1 >= 0 && y1 < H_);
  bool vd = (x1 >= 0 && x1 < W_ && y1 >= 0 && y1 < H_);
  int cx0 = min(max(x0, 0), W_ - 1), cy0 = min(max(y0, 0), H_ - 1);
  int cx1 = min(max(x1, 0), W_ - 1), cy1 = min(max(y1, 0), H_ - 1);
  size_t oa = (size_t)cy0 * W_ + cx0, ob = (size_t)cy0 * W_ + cx1;
  size_t oc = (size_t)cy1 * W_ + cx0, od = (size_t)cy1 * W_ + cx1;
  const float* fl = feats + (size_t)l * C_ * HW_;
  float* ol = out + (size_t)l * C_ * HW_ + (size_t)y * W_ + x;
  for (int c = 0; c < C_; ++c) {
    const float* fc = fl + (size_t)c * HW_;
    if (c + 1 < C_) __builtin_prefetch(fc + HW_ + oa, 0, 1);
    float acc = 0.f;
    if (va) acc = fmaf(fc[oa], wa, acc);
    if (vb) acc = fmaf(fc[ob], wb, acc);
    if (vc) acc = fmaf(fc[oc], wc, acc);
    if (vd) acc = fmaf(fc[od], wd, acc);
    ol[(size_t)c * HW_] = acc;
  }
}

extern "C" void kernel_launch(void* const* d_in, const int* in_sizes, int n_in,
                              void* d_out, int out_size, void* d_ws, size_t ws_size,
                              hipStream_t stream) {
  (void)in_sizes; (void)n_in; (void)out_size; (void)ws_size;
  const float* feats   = (const float*)d_in[0];
  const float* convS_w = (const float*)d_in[1];
  const float* convS_b = (const float*)d_in[2];
  const float* convD_w = (const float*)d_in[3];
  const float* convD_b = (const float*)d_in[4];
  const float* proj_w  = (const float*)d_in[5];
  const float* proj_b  = (const float*)d_in[6];
  const float* projS_w = (const float*)d_in[7];
  const float* projS_b = (const float*)d_in[8];
  const float* final_w = (const float*)d_in[9];
  const float* final_b = (const float*)d_in[10];
  float* out = (float*)d_out;
  float* ws = (float*)d_ws;

  size_t off = 0;
  auto allocF = [&](size_t n) { float* p = ws + off; off += (n + 15) & ~(size_t)15; return p; };
  auto allocB = [&](size_t n) { __bf16* p = (__bf16*)(ws + off);
                                off += (((n + 1) / 2) + 15) & ~(size_t)15; return p; };

  const size_t LHW = (size_t)L_ * HW_;       // 131072
  const size_t LDK = (size_t)L_ * CD_ * KP_; // 131072
  float* S      = allocF(LHW);
  float* T      = allocF(LHW);
  float* P      = allocF(LHW);
  float* P2     = allocF(LHW);
  float* M      = allocF(LHW);
  float* SS     = allocF(LHW);
  float* NM     = allocF(LHW);
  float* DESC   = allocF(LDK);
  float* PROJ   = allocF(LDK);
  float* SOUT   = allocF(LDK);
  float* SORTED = allocF(LDK);
  float* OUT3   = allocF((size_t)L_ * 3 * KP_);
  float* MD     = allocF(16);
  float* THETA  = allocF(32);
  float* ZEROS  = allocF(64);
  int*   TOPK   = (int*)allocF((size_t)L_ * KP_);
  __bf16* WBFD  = allocB(294912);
  __bf16* WBFP  = allocB(49152);
  __bf16* WBFS  = allocB(16384);
  __bf16* D1BF  = allocB(LDK);
  __bf16* PKBF  = allocB(LDK);

  dim3 rowG(L_ * H_), rowB(W_);

  k_repack<<<dim3(1152), dim3(256), 0, stream>>>(convD_w, proj_w, projS_w,
                                                 WBFD, WBFP, WBFS, ZEROS);
  k_scores_conv<<<rowG, rowB, 0, stream>>>(feats, convS_w, convS_b, S);

  // simple_nms
  k_pool_x<<<rowG, rowB, 0, stream>>>(S, T);
  k_pool_y<<<rowG, rowB, 0, stream>>>(T, P);
  k_eq_mask<<<rowG, rowB, 0, stream>>>(S, P, M);
  for (int it = 0; it < 2; ++it) {
    k_pool_x<<<rowG, rowB, 0, stream>>>(M, T);
    k_pool_y<<<rowG, rowB, 0, stream>>>(T, P);
    k_supp_scores<<<rowG, rowB, 0, stream>>>(P, S, SS);
    k_pool_x<<<rowG, rowB, 0, stream>>>(SS, T);
    k_pool_y<<<rowG, rowB, 0, stream>>>(T, P2);
    k_mask_update<<<rowG, rowB, 0, stream>>>(M, SS, P2, P);
  }
  k_apply_mask<<<rowG, rowB, 0, stream>>>(M, S, NM);

  k_topk<<<dim3(L_), dim3(1024), 0, stream>>>(NM, TOPK);

  k_desc_conv<<<dim3(L_ * 16), dim3(256), 0, stream>>>(feats, WBFD, convD_b, TOPK, ZEROS, DESC);
  k_normalize<<<dim3(L_ * KP_), dim3(CD_), 0, stream>>>(DESC);
  k_attention<<<dim3(KP_), dim3(CD_), 0, stream>>>(DESC, D1BF);
  k_proj_conv<<<dim3(L_ * 16), dim3(256), 0, stream>>>(D1BF, WBFP, proj_b, PROJ, PKBF);
  k_projS_conv<<<dim3(L_ * 16), dim3(256), 0, stream>>>(PKBF, WBFS, projS_b, SOUT);
  k_sort<<<dim3(L_ * CD_), dim3(KP_), 0, stream>>>(SOUT, PROJ, SORTED);
  k_final<<<dim3(L_), dim3(KP_), 0, stream>>>(SORTED, final_w, final_b, OUT3);
  k_mindesc<<<dim3(L_ * 3), dim3(KP_), 0, stream>>>(OUT3, MD);
  k_theta<<<dim3(1), dim3(32), 0, stream>>>(MD, THETA);
  k_grid_sample<<<rowG, rowB, 0, stream>>>(feats, THETA, out);
}
